// MoE_29094108463843
// MI455X (gfx1250) — matrix operand order repaired and checked by reading, not compiled, not run on hardware
//
#include <hip/hip_runtime.h>
#include <stdint.h>

#define D_DIM   3072
#define N_EXP   8
#define H_DIM   128
#define O_DIM   10
#define B_TOK   8192
#define CHUNK_K 256
#define TILE_M  32
#define NCHUNK  (D_DIM / CHUNK_K)   // 12

typedef __attribute__((ext_vector_type(16))) __bf16 v16bf;
typedef __attribute__((ext_vector_type(8)))  float  v8f;

union Frag {              // 16 bf16 = 32 bytes, assembled from two 16B chunks
    v16bf v;
    uint4 q[2];
};

// CDNA5 async memory->LDS copy (16B per lane), tracked by ASYNCcnt.
// lds_off = wave-relative LDS byte offset, gptr = global source address.
__device__ __forceinline__ void async_copy_b128(uint32_t lds_off, const void* gptr) {
    asm volatile("global_load_async_to_lds_b128 %0, %1, off"
                 :: "v"(lds_off), "v"(gptr) : "memory");
}
__device__ __forceinline__ void wait_async0() {
    asm volatile("s_wait_asynccnt 0x0" ::: "memory");
}

// ---------------------------------------------------------------------------
// Kernel 1: zero the output accumulator and the per-expert counters
// ---------------------------------------------------------------------------
__global__ void moe_init(float* __restrict__ out, int* __restrict__ cnt) {
    int i = blockIdx.x * blockDim.x + threadIdx.x;
    if (i < B_TOK * O_DIM) out[i] = 0.0f;
    if (i < N_EXP) cnt[i] = 0;
}

// ---------------------------------------------------------------------------
// Kernel 2: convert x fp32 -> bf16 (vectorized, 4 elems/thread)
// ---------------------------------------------------------------------------
__global__ void conv_x(const float* __restrict__ x, uint16_t* __restrict__ xb) {
    size_t i = (size_t)(blockIdx.x * blockDim.x + threadIdx.x) * 4;
    float4 f = *(const float4*)(x + i);
    union { __bf16 b[4]; uint64_t u; } t;
    t.b[0] = (__bf16)f.x; t.b[1] = (__bf16)f.y;
    t.b[2] = (__bf16)f.z; t.b[3] = (__bf16)f.w;
    *(uint64_t*)(xb + i) = t.u;
}

// ---------------------------------------------------------------------------
// Kernel 3: W1 [e][d][h] fp32 -> W1T [e][h][d] bf16  (coalesced writes)
// ---------------------------------------------------------------------------
__global__ void conv_w1(const float* __restrict__ w1, uint16_t* __restrict__ w1t) {
    int i = blockIdx.x * blockDim.x + threadIdx.x;
    if (i >= N_EXP * H_DIM * D_DIM) return;
    int d  = i % D_DIM;
    int eh = i / D_DIM;
    int h  = eh % H_DIM;
    int e  = eh / H_DIM;
    __bf16 v = (__bf16)w1[((size_t)e * D_DIM + d) * H_DIM + h];
    w1t[i] = *(uint16_t*)&v;
}

// ---------------------------------------------------------------------------
// Kernel 4: gating.  One wave32 per token: 8 logits via lane-split dot
// products, shuffle reduction, top-2 softmax, atomic scatter to expert lists.
// ---------------------------------------------------------------------------
__global__ void gate_route(const float* __restrict__ x, const float* __restrict__ wg,
                           const float* __restrict__ bg, int* __restrict__ cnt,
                           int* __restrict__ toks, float* __restrict__ wgts) {
    int wave = threadIdx.x >> 5;
    int lane = threadIdx.x & 31;
    int b = blockIdx.x * 8 + wave;
    const float* xr = x + (size_t)b * D_DIM;

    float acc[N_EXP];
#pragma unroll
    for (int e = 0; e < N_EXP; ++e) acc[e] = 0.0f;

    for (int d = lane; d < D_DIM; d += 32) {
        float xv = xr[d];
        const float* wr = wg + d * N_EXP;
#pragma unroll
        for (int e = 0; e < N_EXP; ++e) acc[e] += xv * wr[e];
    }
#pragma unroll
    for (int off = 16; off >= 1; off >>= 1) {
#pragma unroll
        for (int e = 0; e < N_EXP; ++e) acc[e] += __shfl_down(acc[e], off, 32);
    }
    if (lane == 0) {
        float g[N_EXP];
#pragma unroll
        for (int e = 0; e < N_EXP; ++e) g[e] = acc[e] + bg[e];
        int i0 = 0;
#pragma unroll
        for (int e = 1; e < N_EXP; ++e) if (g[e] > g[i0]) i0 = e;
        int i1 = (i0 == 0) ? 1 : 0;
#pragma unroll
        for (int e = 0; e < N_EXP; ++e) if (e != i0 && g[e] > g[i1]) i1 = e;
        float m  = fmaxf(g[i0], g[i1]);
        float e0 = __expf(g[i0] - m);
        float e1 = __expf(g[i1] - m);
        float inv = 1.0f / (e0 + e1);
        int p0 = atomicAdd(&cnt[i0], 1);
        toks[i0 * B_TOK + p0] = b; wgts[i0 * B_TOK + p0] = e0 * inv;
        int p1 = atomicAdd(&cnt[i1], 1);
        toks[i1 * B_TOK + p1] = b; wgts[i1 * B_TOK + p1] = e1 * inv;
    }
}

// ---------------------------------------------------------------------------
// Kernel 5: fused expert FC1 (bf16 WMMA) + ReLU + FC2 + weighted combine.
// Block = 256 threads = 8 waves; tile = 32 gathered tokens x H=128 outputs.
// Wave w owns M-subtile (w&1)*16 and H-quarter (w>>1)*32 -> two 16x16 accs.
// A chunks are DOUBLE-BUFFERED in LDS via global_load_async_to_lds_b128:
// chunk c+1 streams in (ASYNCcnt) while the 16 WMMAs of chunk c execute.
// Per-thread source pointers / LDS offsets hoisted out of the chunk loop so
// the async issue path is pure pointer arithmetic (no per-chunk ds_load).
// ---------------------------------------------------------------------------
__global__ __launch_bounds__(256) void expert_tile(
        const uint16_t* __restrict__ xb, const uint16_t* __restrict__ w1t,
        const float* __restrict__ b1, const float* __restrict__ w2,
        const float* __restrict__ b2, const int* __restrict__ cnt,
        const int* __restrict__ toks, const float* __restrict__ wgts,
        float* __restrict__ out) {
    int e = blockIdx.y;
    int t = blockIdx.x;
    int nTok = cnt[e];
    if (t * TILE_M >= nTok) return;

    // 32 KB: two 16 KB A chunk buffers; h tile (32x128 f32 = 16 KB) aliases buf0
    __shared__ __align__(16) uint16_t a_lds[2][TILE_M][CHUNK_K];
    float (*h_lds)[H_DIM] = (float(*)[H_DIM])&a_lds[0][0][0];
    __shared__ int   s_tok[TILE_M];
    __shared__ float s_w[TILE_M];

    int tid = threadIdx.x;
    if (tid < TILE_M) {
        int p = t * TILE_M + tid;
        if (p < nTok) { s_tok[tid] = toks[e * B_TOK + p]; s_w[tid] = wgts[e * B_TOK + p]; }
        else          { s_tok[tid] = 0;                   s_w[tid] = 0.0f; }
    }
    __syncthreads();

    int wave   = tid >> 5;
    int lane   = tid & 31;
    int msub   = wave & 1;       // 0..1 : rows msub*16 .. +15
    int hq     = wave >> 1;      // 0..3 : cols hq*32  .. +31
    int laneHi = lane >> 4;      // ISA 7.12.2 half-wave split
    int laneLo = lane & 15;

    v8f acc0 = {}, acc1 = {};
    const uint16_t* w1e = w1t + (size_t)e * H_DIM * D_DIM;
    uint32_t a_base = (uint32_t)(size_t)&a_lds[0][0][0];   // wave-rel LDS byte off

    // hoist this thread's 4 staging descriptors (fixed row/col for whole tile)
    const uint16_t* src[4];
    uint32_t        loff[4];
#pragma unroll
    for (int i = 0; i < 4; ++i) {
        int f  = tid + i * 256;            // 0..1023 flat b128 index
        int r  = f >> 5;                   // 32 x 16B per row
        int kc = (f & 31) << 3;            // bf16 element offset
        src[i]  = &xb[(size_t)s_tok[r] * D_DIM + kc];
        loff[i] = a_base + (uint32_t)((r * CHUNK_K + kc) * 2);
    }

    // issue async stage of chunk 0 into buffer 0 (4 x 16B per thread)
#pragma unroll
    for (int i = 0; i < 4; ++i) async_copy_b128(loff[i], src[i]);

    for (int c = 0; c < NCHUNK; ++c) {
        int buf = c & 1;
        wait_async0();          // my copies for chunk c are in LDS
        __syncthreads();        // everyone's copies done; prev compute done
        if (c + 1 < NCHUNK) {   // stream chunk c+1 into the other buffer
            uint32_t boff = (uint32_t)((buf ^ 1) * TILE_M * CHUNK_K * 2);
            int      goff = (c + 1) * CHUNK_K;
#pragma unroll
            for (int i = 0; i < 4; ++i)
                async_copy_b128(loff[i] + boff, src[i] + goff);
        }
#pragma unroll
        for (int kk = 0; kk < CHUNK_K; kk += 32) {
            // A fragment: lane<16 holds K 0-15, lane>=16 holds 16-31
            Frag a;
            int arow = msub * 16 + laneLo;
            a.q[0] = *(const uint4*)&a_lds[buf][arow][kk +     laneHi * 16];
            a.q[1] = *(const uint4*)&a_lds[buf][arow][kk + 8 + laneHi * 16];
            // B fragments: lane = column, contiguous 16 K values (W1 transposed)
            int gk   = c * CHUNK_K + kk + laneHi * 16;
            int col0 = hq * 32 + laneLo;
            v16bf bv0 = *(const v16bf*)&w1e[(size_t)col0 * D_DIM + gk];
            v16bf bv1 = *(const v16bf*)&w1e[(size_t)(col0 + 16) * D_DIM + gk];
            acc0 = __builtin_amdgcn_wmma_f32_16x16x32_bf16(
                       false, a.v, false, bv0, (short)0, acc0, false, false);
            acc1 = __builtin_amdgcn_wmma_f32_16x16x32_bf16(
                       false, a.v, false, bv1, (short)0, acc1, false, false);
        }
        __syncthreads();        // all waves done with buf before it is reused
    }

    // bias + ReLU, spill h tile to LDS (aliases buf0; all WMMAs complete)
    {
        int h0  = hq * 32;
        float bb0 = b1[e * H_DIM + h0 + laneLo];
        float bb1 = b1[e * H_DIM + h0 + 16 + laneLo];
#pragma unroll
        for (int r = 0; r < 8; ++r) {
            int m = msub * 16 + r + laneHi * 8;   // C layout: vgpr r, lane half
            h_lds[m][h0 + laneLo]      = fmaxf(acc0[r] + bb0, 0.0f);
            h_lds[m][h0 + 16 + laneLo] = fmaxf(acc1[r] + bb1, 0.0f);
        }
    }
    __syncthreads();

    // FC2 (128 -> 10) + gate-weighted combine with global atomics
    const float* w2e = w2 + (size_t)e * H_DIM * O_DIM;
    const float* b2e = b2 + (size_t)e * O_DIM;
    for (int i = tid; i < TILE_M * O_DIM; i += 256) {
        int m = i / O_DIM, o = i % O_DIM;
        float y = b2e[o];
#pragma unroll 8
        for (int h = 0; h < H_DIM; ++h) y += h_lds[m][h] * w2e[h * O_DIM + o];
        float w = s_w[m];
        if (w != 0.0f) atomicAdd(&out[(size_t)s_tok[m] * O_DIM + o], w * y);
    }
}

// ---------------------------------------------------------------------------
extern "C" void kernel_launch(void* const* d_in, const int* in_sizes, int n_in,
                              void* d_out, int out_size, void* d_ws, size_t ws_size,
                              hipStream_t stream) {
    const float* x  = (const float*)d_in[0];
    const float* Wg = (const float*)d_in[1];
    const float* bg = (const float*)d_in[2];
    const float* W1 = (const float*)d_in[3];
    const float* b1 = (const float*)d_in[4];
    const float* W2 = (const float*)d_in[5];
    const float* b2 = (const float*)d_in[6];
    float* out = (float*)d_out;

    char* ws = (char*)d_ws;
    size_t off = 0;
    uint16_t* w1t  = (uint16_t*)(ws + off); off += (size_t)N_EXP * H_DIM * D_DIM * 2;
    uint16_t* xb   = (uint16_t*)(ws + off); off += (size_t)B_TOK * D_DIM * 2;
    int*      cnt  = (int*)(ws + off);      off += 256;
    int*      toks = (int*)(ws + off);      off += (size_t)N_EXP * B_TOK * 4;
    float*    wgts = (float*)(ws + off);    off += (size_t)N_EXP * B_TOK * 4;

    moe_init  <<<(B_TOK * O_DIM + 255) / 256, 256, 0, stream>>>(out, cnt);
    conv_x    <<<(B_TOK * D_DIM / 4) / 256,   256, 0, stream>>>(x, xb);
    conv_w1   <<<(N_EXP * H_DIM * D_DIM + 255) / 256, 256, 0, stream>>>(W1, w1t);
    gate_route<<<B_TOK / 8, 256, 0, stream>>>(x, Wg, bg, cnt, toks, wgts);
    expert_tile<<<dim3(B_TOK / TILE_M, N_EXP), 256, 0, stream>>>(
        xb, w1t, b1, W2, b2, cnt, toks, wgts, out);
}
